// VisionTokenMerger_81956565942277
// MI455X (gfx1250) — compile-verified, matrix operand order checked by reading, not run on hardware
//
#include <hip/hip_runtime.h>

// ---------------------------------------------------------------------------
// VisionTokenMerger for MI455X (gfx1250, wave32)
//   K: (32, 1152, 1024) f32
//   out: merged (32, 288, 1024) f32  ++  scalar 576
// Pipeline:
//   1) normalize rows, downconvert to bf16 (set1n even tokens, set2n odd)
//   2) sim[b] = set1n[b] @ set2n[b]^T  via v_wmma_f32_16x16x32_bf16
//      (LDS double-buffered, async global->LDS staging when available)
//   3) per-batch radix-select top-288 threshold + ordered compaction
//   4) gather pairs from original f32 K, average, store
// ---------------------------------------------------------------------------

typedef __attribute__((ext_vector_type(16))) __bf16 v16bf;
typedef __attribute__((ext_vector_type(8)))  float  v8f;
typedef __attribute__((ext_vector_type(4)))  int    v4i;
typedef unsigned short ushort_t;
typedef unsigned int   uint_t;

#define BATCH   32
#define NTOK    1152
#define HID     1024
#define M_TOK   576            // tokens per set
#define R_TOP   288
#define SIM_N   (M_TOK * M_TOK) // 331776

// ---------------- workspace layout (bytes) ----------------
#define S1N_OFF   ((size_t)0)
#define SET_BYTES ((size_t)BATCH * M_TOK * HID * 2)          // 37,748,736
#define S2N_OFF   (S1N_OFF + SET_BYTES)
#define SIM_OFF   (S2N_OFF + SET_BYTES)
#define SIM_BYTES ((size_t)BATCH * SIM_N * 4)                // 42,467,328
#define PAIR_OFF  (SIM_OFF + SIM_BYTES)

#ifndef __has_builtin
#define __has_builtin(x) 0
#endif
#if __has_builtin(__builtin_amdgcn_global_load_async_to_lds_b128) && \
    __has_builtin(__builtin_amdgcn_s_wait_asynccnt)
#define USE_ASYNC_LDS 1
#else
#define USE_ASYNC_LDS 0
#endif

// round-to-nearest-even f32 -> bf16 bits
__device__ __forceinline__ ushort_t f2bf(float f) {
  uint_t u = __float_as_uint(f);
  u += 0x7FFFu + ((u >> 16) & 1u);
  return (ushort_t)(u >> 16);
}

// order-preserving float -> uint key (greater value <=> greater key)
__device__ __forceinline__ uint_t fkey(float f) {
  uint_t u = __float_as_uint(f);
  return (u & 0x80000000u) ? ~u : (u | 0x80000000u);
}

// ---------------------------------------------------------------------------
// Kernel 1: per-token L2 norm, write normalized bf16 rows.
// grid = 32*1152 blocks, 256 threads; each thread owns one float4 (4 elems).
// ---------------------------------------------------------------------------
__global__ void norm_bf16_kernel(const float* __restrict__ K,
                                 ushort_t* __restrict__ s1n,
                                 ushort_t* __restrict__ s2n) {
  const int token = blockIdx.x;            // 0 .. 36863
  const int b = token / NTOK;
  const int t = token - b * NTOK;
  const int tid = threadIdx.x;             // 0..255

  const float4 v = ((const float4*)(K + (size_t)token * HID))[tid];

  __shared__ float red[256];
  __shared__ float s_inv;
  red[tid] = v.x * v.x + v.y * v.y + v.z * v.z + v.w * v.w;
  __syncthreads();
  for (int off = 128; off > 0; off >>= 1) {
    if (tid < off) red[tid] += red[tid + off];
    __syncthreads();
  }
  if (tid == 0) {
    float n = sqrtf(red[0]);
    s_inv = 1.0f / fmaxf(n, 1e-12f);
  }
  __syncthreads();
  const float inv = s_inv;

  ushort_t* dst = ((t & 1) ? s2n : s1n) +
                  ((size_t)b * M_TOK + (size_t)(t >> 1)) * HID + (size_t)tid * 4;
  ushort4 o;
  o.x = f2bf(v.x * inv);
  o.y = f2bf(v.y * inv);
  o.z = f2bf(v.z * inv);
  o.w = f2bf(v.w * inv);
  *(ushort4*)dst = o;
}

// ---------------------------------------------------------------------------
// Kernel 2: sim[b] = s1n[b] (576x1024) @ s2n[b]^T  -> (576x576) f32
// Block = 128 threads (4 waves). Block tile 64(M) x 64(N).
// Wave w: wm = w&1 (32-row stripe), wn = w>>1 (32-col stripe).
// Each wave: 2 A frags x 2 B frags -> 4 WMMAs per 32-wide k-step.
// K staged through double-buffered LDS (async global->LDS if available).
// ---------------------------------------------------------------------------
union Frag {
  uint4 q[2];
  v16bf v;
};

__device__ __forceinline__ void async_cp16(const ushort_t* g, ushort_t* l) {
#if USE_ASYNC_LDS
  __builtin_amdgcn_global_load_async_to_lds_b128(
      (__attribute__((address_space(1))) v4i*)const_cast<ushort_t*>(g),
      (__attribute__((address_space(3))) v4i*)l,
      0, 0);
#else
  (void)g; (void)l;
#endif
}

__global__ void sim_wmma_kernel(const ushort_t* __restrict__ s1n,
                                const ushort_t* __restrict__ s2n,
                                float* __restrict__ sim) {
  const int b  = blockIdx.z;
  const int bm = blockIdx.y;   // 0..8
  const int bn = blockIdx.x;   // 0..8

  __shared__ ushort_t lA[2][64 * 32];
  __shared__ ushort_t lB[2][64 * 32];

  const int tid  = threadIdx.x;   // 0..127
  const int lane = tid & 31;
  const int wave = tid >> 5;      // 0..3
  const int wm   = wave & 1;      // 32-row stripe
  const int wn   = wave >> 1;     // 32-col stripe

  // staging: thread t covers row r = t>>1, 16 halves at col (t&1)*16 (2 x b128)
  const int r  = tid >> 1;        // 0..63
  const int cc = (tid & 1) * 16;  // 0 or 16
  const ushort_t* gA = s1n + ((size_t)b * M_TOK + (size_t)(bm * 64 + r)) * HID + cc;
  const ushort_t* gB = s2n + ((size_t)b * M_TOK + (size_t)(bn * 64 + r)) * HID + cc;

  v8f c00 = {}, c01 = {}, c10 = {}, c11 = {};

  const int klo   = (lane < 16) ? 0 : 8;
  const int mrow0 = wm * 32 + (lane & 15);
  const int mrow1 = mrow0 + 16;
  const int nrow0 = wn * 32 + (lane & 15);
  const int nrow1 = nrow0 + 16;

  auto compute = [&](int buf) {
    const ushort_t* A = lA[buf];
    const ushort_t* B = lB[buf];
    Frag a0, a1, fb0, fb1;
    a0.q[0]  = *(const uint4*)&A[mrow0 * 32 + klo];
    a0.q[1]  = *(const uint4*)&A[mrow0 * 32 + klo + 16];
    a1.q[0]  = *(const uint4*)&A[mrow1 * 32 + klo];
    a1.q[1]  = *(const uint4*)&A[mrow1 * 32 + klo + 16];
    fb0.q[0] = *(const uint4*)&B[nrow0 * 32 + klo];
    fb0.q[1] = *(const uint4*)&B[nrow0 * 32 + klo + 16];
    fb1.q[0] = *(const uint4*)&B[nrow1 * 32 + klo];
    fb1.q[1] = *(const uint4*)&B[nrow1 * 32 + klo + 16];
    c00 = __builtin_amdgcn_wmma_f32_16x16x32_bf16(false, a0.v, false, fb0.v,
                                                  (short)0, c00, false, false);
    c01 = __builtin_amdgcn_wmma_f32_16x16x32_bf16(false, a0.v, false, fb1.v,
                                                  (short)0, c01, false, false);
    c10 = __builtin_amdgcn_wmma_f32_16x16x32_bf16(false, a1.v, false, fb0.v,
                                                  (short)0, c10, false, false);
    c11 = __builtin_amdgcn_wmma_f32_16x16x32_bf16(false, a1.v, false, fb1.v,
                                                  (short)0, c11, false, false);
  };

#if USE_ASYNC_LDS
  auto stage_async = [&](int k, int buf) {
    async_cp16(gA + k,     &lA[buf][r * 32 + cc]);
    async_cp16(gA + k + 8, &lA[buf][r * 32 + cc + 8]);
    async_cp16(gB + k,     &lB[buf][r * 32 + cc]);
    async_cp16(gB + k + 8, &lB[buf][r * 32 + cc + 8]);
  };

  stage_async(0, 0);
  for (int k0 = 0; k0 < HID; k0 += 32) {
    const int cur = (k0 >> 5) & 1;
    __builtin_amdgcn_s_wait_asynccnt(0);   // my async loads for buf 'cur' done
    __syncthreads();                        // all waves' loads visible in LDS
    if (k0 + 32 < HID) stage_async(k0 + 32, cur ^ 1);  // overlap with WMMAs
    compute(cur);
    __syncthreads();                        // everyone done reading 'cur'
  }
#else
  auto stage_sync = [&](int k, int buf) {
    uint4 a0 = *(const uint4*)(gA + k);
    uint4 a1 = *(const uint4*)(gA + k + 8);
    uint4 b0 = *(const uint4*)(gB + k);
    uint4 b1 = *(const uint4*)(gB + k + 8);
    *(uint4*)&lA[buf][r * 32 + cc]     = a0;
    *(uint4*)&lA[buf][r * 32 + cc + 8] = a1;
    *(uint4*)&lB[buf][r * 32 + cc]     = b0;
    *(uint4*)&lB[buf][r * 32 + cc + 8] = b1;
  };

  for (int k0 = 0; k0 < HID; k0 += 32) {
    stage_sync(k0, 0);
    if (k0 + 32 < HID) {
      __builtin_prefetch((const void*)(gA + k0 + 32), 0, 1);
      __builtin_prefetch((const void*)(gB + k0 + 32), 0, 1);
    }
    __syncthreads();
    compute(0);
    __syncthreads();
  }
#endif

  // C/D layout: VGPR v, lanes 0-15 -> M=v, lanes 16-31 -> M=v+8; N = lane%16
  const int n     = lane & 15;
  const int mbase = (lane < 16) ? 0 : 8;
  float* out0 = sim + (size_t)b * SIM_N +
                (size_t)(bm * 64 + wm * 32 + mbase) * M_TOK +
                (bn * 64 + wn * 32 + n);
  float* out1 = out0 + (size_t)16 * M_TOK;   // rows +16 (a1 results)
  #pragma unroll
  for (int v = 0; v < 8; ++v) {
    out0[(size_t)v * M_TOK]      = c00[v];
    out0[(size_t)v * M_TOK + 16] = c01[v];
    out1[(size_t)v * M_TOK]      = c10[v];
    out1[(size_t)v * M_TOK + 16] = c11[v];
  }
}

// ---------------------------------------------------------------------------
// Kernel 3: deterministic top-288 per batch.
//  Phase A: 4-pass radix select (8-bit digits, MSB first) over fkey() to
//           find threshold key T, count G of keys > T, tie quota = R - G.
//  Phase B: index-ordered block-scan compaction: take all keys > T plus the
//           first (R-G) keys == T in flat-index order. Deterministic.
// One block (1024 threads) per batch.
// ---------------------------------------------------------------------------
__device__ __forceinline__ void inclusive_scan_1024(uint_t* s) {
  const int tid = threadIdx.x;
  for (int off = 1; off < 1024; off <<= 1) {
    uint_t v = 0;
    if (tid >= off) v = s[tid - off];
    __syncthreads();
    s[tid] += v;
    __syncthreads();
  }
}

__global__ void topk_select_kernel(const float* __restrict__ sim,
                                   int2* __restrict__ pairs) {
  const int b   = blockIdx.x;
  const int tid = threadIdx.x;
  const float* simb = sim + (size_t)b * SIM_N;

  __shared__ uint_t hist[256];
  __shared__ uint_t s_prefix, s_greater, s_remain;
  __shared__ uint_t scanT[1024];
  __shared__ uint_t scanS[1024];
  __shared__ uint_t baseSel, baseTie;

  if (tid == 0) { s_prefix = 0u; s_greater = 0u; s_remain = R_TOP; }
  __syncthreads();

  // ---- Phase A: radix select ----
  for (int pass = 0; pass < 4; ++pass) {
    const int shift = 24 - 8 * pass;
    if (tid < 256) hist[tid] = 0u;
    __syncthreads();
    const uint_t pfx = s_prefix;
    for (int e = tid; e < SIM_N; e += 1024) {
      const uint_t u = fkey(simb[e]);
      bool in_bucket = (pass == 0) || ((u >> (shift + 8)) == pfx);
      if (in_bucket) atomicAdd(&hist[(u >> shift) & 255u], 1u);
    }
    __syncthreads();
    if (tid == 0) {
      const uint_t rem = s_remain;
      uint_t cum = 0u;
      int d = 255;
      for (; d >= 0; --d) {
        const uint_t h = hist[d];
        if (cum + h >= rem) break;
        cum += h;
      }
      if (d < 0) d = 0;
      s_prefix  = (pfx << 8) | (uint_t)d;
      s_greater = s_greater + cum;
      s_remain  = rem - cum;
    }
    __syncthreads();
  }

  const uint_t T     = s_prefix;   // full 32-bit threshold key
  const uint_t quota = s_remain;   // number of ties (== T) to take

  if (tid == 0) { baseSel = 0u; baseTie = 0u; }
  __syncthreads();

  // ---- Phase B: ordered compaction ----
  for (int c0 = 0; c0 < SIM_N; c0 += 1024) {
    if (baseSel >= R_TOP) break;   // uniform (read after barrier)
    const int e = c0 + tid;
    uint_t u = 0u;
    if (e < SIM_N) u = fkey(simb[e]);
    const uint_t isG = (e < SIM_N && u > T)  ? 1u : 0u;
    const uint_t isT = (e < SIM_N && u == T) ? 1u : 0u;

    scanT[tid] = isT;
    __syncthreads();
    inclusive_scan_1024(scanT);
    const uint_t tieExcl = scanT[tid] - isT;
    const uint_t takeTie = (isT && (baseTie + tieExcl) < quota) ? 1u : 0u;
    const uint_t sel = isG | takeTie;

    scanS[tid] = sel;
    __syncthreads();
    inclusive_scan_1024(scanS);
    const uint_t selExcl = scanS[tid] - sel;

    if (sel) {
      const uint_t slot = baseSel + selExcl;
      if (slot < R_TOP) {
        int2 p;
        p.x = e / M_TOK;   // i (set1 index)
        p.y = e % M_TOK;   // j (set2 index)
        pairs[(size_t)b * R_TOP + slot] = p;
      }
    }
    __syncthreads();
    if (tid == 0) {
      baseSel += scanS[1023];
      baseTie += scanT[1023];
    }
    __syncthreads();
  }
}

// ---------------------------------------------------------------------------
// Kernel 4: merged[b][r] = (K[b][2i] + K[b][2j+1]) * 0.5, plus scalar 576.
// grid (288, 32), 256 threads; each thread one float4.
// ---------------------------------------------------------------------------
__global__ void merge_gather_kernel(const float* __restrict__ K,
                                    const int2* __restrict__ pairs,
                                    float* __restrict__ out) {
  const int r = blockIdx.x;   // 0..287
  const int b = blockIdx.y;   // 0..31
  const int tid = threadIdx.x;

  const int2 p = pairs[(size_t)b * R_TOP + r];
  const float4* t1 =
      (const float4*)(K + ((size_t)b * NTOK + (size_t)(2 * p.x)) * HID);
  const float4* t2 =
      (const float4*)(K + ((size_t)b * NTOK + (size_t)(2 * p.y + 1)) * HID);
  float4* o = (float4*)(out + ((size_t)b * R_TOP + r) * HID);

  const float4 a = t1[tid];
  const float4 c = t2[tid];
  float4 m;
  m.x = (a.x + c.x) * 0.5f;
  m.y = (a.y + c.y) * 0.5f;
  m.z = (a.z + c.z) * 0.5f;
  m.w = (a.w + c.w) * 0.5f;
  o[tid] = m;

  if (b == 0 && r == 0 && tid == 0) {
    out[(size_t)BATCH * R_TOP * HID] = (float)(NTOK / 2);  // scalar 576
  }
}

// ---------------------------------------------------------------------------
extern "C" void kernel_launch(void* const* d_in, const int* in_sizes, int n_in,
                              void* d_out, int out_size, void* d_ws, size_t ws_size,
                              hipStream_t stream) {
  (void)in_sizes; (void)n_in; (void)out_size; (void)ws_size;

  const float* K = (const float*)d_in[0];
  float* out = (float*)d_out;

  char* ws = (char*)d_ws;
  ushort_t* s1n  = (ushort_t*)(ws + S1N_OFF);
  ushort_t* s2n  = (ushort_t*)(ws + S2N_OFF);
  float*    sim  = (float*)   (ws + SIM_OFF);
  int2*     pair = (int2*)    (ws + PAIR_OFF);

  // 1) normalize + bf16 downconvert
  norm_bf16_kernel<<<dim3(BATCH * NTOK), dim3(256), 0, stream>>>(K, s1n, s2n);

  // 2) batched cosine-sim GEMM via WMMA (4 waves, 32x32 wave tiles)
  sim_wmma_kernel<<<dim3(M_TOK / 64, M_TOK / 64, BATCH), dim3(128), 0, stream>>>(
      s1n, s2n, sim);

  // 3) top-288 per batch (radix select + ordered compaction)
  topk_select_kernel<<<dim3(BATCH), dim3(1024), 0, stream>>>(sim, pair);

  // 4) gather + average merged tokens, emit scalar
  merge_gather_kernel<<<dim3(R_TOP, BATCH), dim3(256), 0, stream>>>(K, pair, out);
}
